// NEP_56934086476143
// MI455X (gfx1250) — matrix operand order repaired and checked by reading, not compiled, not run on hardware
//
#include <hip/hip_runtime.h>

typedef __attribute__((ext_vector_type(16))) _Float16 v16h;
typedef __attribute__((ext_vector_type(8)))  _Float16 v8h;
typedef __attribute__((ext_vector_type(8)))  float    v8f;

#define N_ATOMS 8192
#define M_NBR   100
#define MA_NBR  40
#define NP      5      // NMAX+1 (radial and angular)
#define NK      9      // NBASE+1
#define LMAXV   4
#define FEATN   25     // NP + NP*LMAX
#define KPAD    32     // K padded for wmma 16x16x32
#define H_DIM   100
#define HPAD    112    // 7 tiles of 16
#define NTILES  (HPAD / 16)
#define RC_RV   8.0f
#define RC_AV   4.0f
#define PI_F    3.14159265358979f

// ---------------------------------------------------------------------------
// Kernel 1: per-atom feature construction. One wave32 per atom, 8 atoms/block.
// ---------------------------------------------------------------------------
__global__ __launch_bounds__(256) void nep_feat_kernel(
    const float* __restrict__ rij, const float* __restrict__ unitv,
    const int*  __restrict__ itype, const int* __restrict__ jtype,
    const float* __restrict__ c2,  const float* __restrict__ c3,
    const float* __restrict__ q_scaler, _Float16* __restrict__ featH)
{
  __shared__ float s_c2[2*2*NP*NK];          // 180
  __shared__ float s_c3[2*2*NP*NK];          // 180
  __shared__ float s_gna[8][MA_NBR][NP];
  __shared__ float s_u[8][MA_NBR][3];

  const int tid = threadIdx.x;
  if (tid < 2*2*NP*NK) { s_c2[tid] = c2[tid]; s_c3[tid] = c3[tid]; }
  __syncthreads();

  const int wave = tid >> 5;
  const int lane = tid & 31;
  const int n = blockIdx.x * 8 + wave;
  const int it = itype[n];

  // Prefetch the unit-vector row (consumed in the angular phase below).
  __builtin_prefetch(&unitv[(n * M_NBR + lane) * 3], 0, 0);

  // ---- radial q2 -----------------------------------------------------------
  float q2[NP] = {0.f, 0.f, 0.f, 0.f, 0.f};
  for (int m = lane; m < M_NBR; m += 32) {
    const float r  = rij[n * M_NBR + m];
    const int   jt = jtype[n * M_NBR + m];
    const float fc = (r < RC_RV) ? 0.5f * (cosf(PI_F * r * (1.0f / RC_RV)) + 1.0f) : 0.0f;
    const float t  = r * (1.0f / RC_RV) - 1.0f;
    const float x  = 2.0f * t * t - 1.0f;
    float fk[NK];
    float Tm2 = 1.0f, Tm1 = x;
    fk[0] = 0.5f * (Tm2 + 1.0f) * fc;
    fk[1] = 0.5f * (Tm1 + 1.0f) * fc;
    #pragma unroll
    for (int k = 2; k < NK; ++k) {
      const float Tk = 2.0f * x * Tm1 - Tm2;
      fk[k] = 0.5f * (Tk + 1.0f) * fc;
      Tm2 = Tm1; Tm1 = Tk;
    }
    const float* cb = &s_c2[(it * 2 + jt) * NP * NK];
    #pragma unroll
    for (int p = 0; p < NP; ++p) {
      float s = 0.f;
      #pragma unroll
      for (int k = 0; k < NK; ++k) s += fk[k] * cb[p * NK + k];
      q2[p] += s;
    }
  }

  // ---- angular basis gna + unit vectors into LDS ---------------------------
  for (int m = lane; m < MA_NBR; m += 32) {
    const float r  = rij[n * M_NBR + m];
    const int   jt = jtype[n * M_NBR + m];
    const float fc = (r < RC_AV) ? 0.5f * (cosf(PI_F * r * (1.0f / RC_AV)) + 1.0f) : 0.0f;
    const float t  = r * (1.0f / RC_AV) - 1.0f;
    const float x  = 2.0f * t * t - 1.0f;
    float fk[NK];
    float Tm2 = 1.0f, Tm1 = x;
    fk[0] = 0.5f * (Tm2 + 1.0f) * fc;
    fk[1] = 0.5f * (Tm1 + 1.0f) * fc;
    #pragma unroll
    for (int k = 2; k < NK; ++k) {
      const float Tk = 2.0f * x * Tm1 - Tm2;
      fk[k] = 0.5f * (Tk + 1.0f) * fc;
      Tm2 = Tm1; Tm1 = Tk;
    }
    const float* cb = &s_c3[(it * 2 + jt) * NP * NK];
    #pragma unroll
    for (int a = 0; a < NP; ++a) {
      float s = 0.f;
      #pragma unroll
      for (int k = 0; k < NK; ++k) s += fk[k] * cb[a * NK + k];
      s_gna[wave][m][a] = s;
    }
    s_u[wave][m][0] = unitv[(n * M_NBR + m) * 3 + 0];
    s_u[wave][m][1] = unitv[(n * M_NBR + m) * 3 + 1];
    s_u[wave][m][2] = unitv[(n * M_NBR + m) * 3 + 2];
  }
  // LDS region is per-wave private; drain DS stores before cross-lane reads.
  asm volatile("s_wait_dscnt 0" ::: "memory");

  // ---- 40x40 pairwise Legendre contraction: 50 pairs per lane --------------
  float acc[NP][LMAXV];
  #pragma unroll
  for (int a = 0; a < NP; ++a)
    #pragma unroll
    for (int l = 0; l < LMAXV; ++l) acc[a][l] = 0.f;

  for (int t = lane; t < MA_NBR * MA_NBR; t += 32) {
    const int j = t / MA_NBR;
    const int k = t - j * MA_NBR;
    const float cjk = s_u[wave][j][0] * s_u[wave][k][0]
                    + s_u[wave][j][1] * s_u[wave][k][1]
                    + s_u[wave][j][2] * s_u[wave][k][2];
    const float c2_ = cjk * cjk;
    const float c3_ = c2_ * cjk;
    const float c4_ = c2_ * c2_;
    float P[LMAXV];
    P[0] = cjk;
    P[1] = 1.5f * c2_ - 0.5f;
    P[2] = 2.5f * c3_ - 1.5f * cjk;
    P[3] = 0.125f * (35.0f * c4_ - 30.0f * c2_ + 3.0f);
    #pragma unroll
    for (int a = 0; a < NP; ++a) {
      const float g = s_gna[wave][j][a] * s_gna[wave][k][a];
      #pragma unroll
      for (int l = 0; l < LMAXV; ++l) acc[a][l] += g * P[l];
    }
  }

  // ---- wave reduction of 25 feature values ---------------------------------
  float vals[FEATN];
  #pragma unroll
  for (int p = 0; p < NP; ++p) vals[p] = q2[p];
  #pragma unroll
  for (int a = 0; a < NP; ++a)
    #pragma unroll
    for (int l = 0; l < LMAXV; ++l) vals[NP + a * LMAXV + l] = acc[a][l];

  #pragma unroll
  for (int f = 0; f < FEATN; ++f) {
    float v = vals[f];
    v += __shfl_down(v, 16, 32);
    v += __shfl_down(v, 8, 32);
    v += __shfl_down(v, 4, 32);
    v += __shfl_down(v, 2, 32);
    v += __shfl_down(v, 1, 32);
    vals[f] = v;
  }

  if (lane == 0) {
    #pragma unroll
    for (int f = 0; f < FEATN; ++f)
      featH[n * KPAD + f] = (_Float16)(vals[f] * q_scaler[f]);
    #pragma unroll
    for (int f = FEATN; f < KPAD; ++f)
      featH[n * KPAD + f] = (_Float16)0.0f;
  }
}

// ---------------------------------------------------------------------------
// Kernel 2: parameter repack.
//  - W0 (T,25,100) f32 -> W0p f16 in WMMA-B fragment order:
//      W0p[((t*NTILES + nt)*32 + lane)*16 + i] = W0[t][K][col]
//      with K = (i<8 ? (lane>>4)*8 + i : 16 + (lane>>4)*8 + (i-8)), col = nt*16 + (lane&15); zero-padded.
//  - b0 (T,100) -> b0p (T,112) f32 zero-padded
//  - W1 (T,100) -> W1p (T,112) f32 zero-padded (pad 0 => padded cols contribute 0)
// ---------------------------------------------------------------------------
__global__ void nep_repack_kernel(const float* __restrict__ W0,
                                  const float* __restrict__ b0,
                                  const float* __restrict__ W1,
                                  _Float16* __restrict__ W0p,
                                  float* __restrict__ b0p,
                                  float* __restrict__ W1p)
{
  const int idx = blockIdx.x * 256 + threadIdx.x;
  if (idx < 2 * NTILES * 32 * 16) {
    const int t    = idx / (NTILES * 32 * 16);
    int rem        = idx - t * (NTILES * 32 * 16);
    const int nt   = rem / (32 * 16);
    rem           -= nt * (32 * 16);
    const int lane = rem >> 4;
    const int i    = rem & 15;
    const int K    = (i < 8) ? (lane >> 4) * 8 + i : 16 + (lane >> 4) * 8 + (i - 8);
    const int col  = nt * 16 + (lane & 15);
    const float v  = (K < FEATN && col < H_DIM) ? W0[(t * FEATN + K) * H_DIM + col] : 0.0f;
    W0p[idx] = (_Float16)v;
  }
  if (idx < 2 * HPAD) {
    const int t   = idx / HPAD;
    const int col = idx - t * HPAD;
    b0p[idx] = (col < H_DIM) ? b0[t * H_DIM + col] : 0.0f;
    W1p[idx] = (col < H_DIM) ? W1[t * H_DIM + col] : 0.0f;
  }
}

// ---------------------------------------------------------------------------
// Kernel 3: MLP via v_wmma_f32_16x16x32_f16. One wave per 16-atom tile.
// Both type-GEMMs computed; selected per atom pre-tanh. Branchless hot loop
// (EXEC stays all-ones through the WMMA region).
// ---------------------------------------------------------------------------
__global__ __launch_bounds__(256) void nep_mlp_kernel(
    const _Float16* __restrict__ featH, const _Float16* __restrict__ W0p,
    const int* __restrict__ itype, const float* __restrict__ b0p,
    const float* __restrict__ W1p, const float* __restrict__ b1,
    float* __restrict__ out)
{
  const int tid  = threadIdx.x;
  const int lane = tid & 31;
  const int wave = tid >> 5;
  const int tile = blockIdx.x * 8 + wave;
  const int base = tile * 16;
  const int half = lane >> 4;          // 0: lanes 0-15, 1: lanes 16-31
  const int l15  = lane & 15;

  // A fragment (16-bit A 16x32 layout): per lane, two contiguous 8-half runs:
  //   K = half*8 + 0..7   and   K = 16 + half*8 + 0..7
  const _Float16* fr = featH + (base + l15) * KPAD + half * 8;
  const v8h alo = *(const v8h*)(fr);
  const v8h ahi = *(const v8h*)(fr + 16);
  const v16h afrag = __builtin_shufflevector(alo, ahi,
      0, 1, 2, 3, 4, 5, 6, 7, 8, 9, 10, 11, 12, 13, 14, 15);

  int   itr[8];
  float eacc[8];
  #pragma unroll
  for (int r = 0; r < 8; ++r) {
    itr[r]  = itype[base + r + half * 8];
    eacc[r] = 0.0f;
  }

  for (int nt = 0; nt < NTILES; ++nt) {
    // B fragments: pre-packed per-lane contiguous 16 halfs (2x b128 loads).
    const v16h bt0 = *(const v16h*)(W0p + ((0 * NTILES + nt) * 32 + lane) * 16);
    const v16h bt1 = *(const v16h*)(W0p + ((1 * NTILES + nt) * 32 + lane) * 16);

    v8f cz = {};
    v8f c0 = __builtin_amdgcn_wmma_f32_16x16x32_f16(
        false, afrag, false, bt0, (short)0, cz, false, false);
    v8f c1 = __builtin_amdgcn_wmma_f32_16x16x32_f16(
        false, afrag, false, bt1, (short)0, cz, false, false);

    const int col = nt * 16 + l15;
    const float bb0 = b0p[col], bb1 = b0p[HPAD + col];
    const float ww0 = W1p[col], ww1 = W1p[HPAD + col];

    // C layout: VGPR r <-> atom row (r + half*8), column = l15.
    #pragma unroll
    for (int r = 0; r < 8; ++r) {
      const bool t1 = (itr[r] != 0);
      const float z  = (t1 ? c1[r] : c0[r]) + (t1 ? bb1 : bb0);
      // branchless tanh: tanh(x) = 1 - 2*rcp(exp(2x)+1); saturates at +-1.
      // v_exp_f32 + v_rcp_f32 only (no IEEE div fixup sequence).
      const float e  = __expf(2.0f * z);
      const float h  = 1.0f - 2.0f * __builtin_amdgcn_rcpf(e + 1.0f);
      float contrib = h * (t1 ? ww1 : ww0);
      contrib += __shfl_xor(contrib, 1, 32);
      contrib += __shfl_xor(contrib, 2, 32);
      contrib += __shfl_xor(contrib, 4, 32);
      contrib += __shfl_xor(contrib, 8, 32);
      eacc[r] += contrib;
    }
  }

  if (l15 == 0) {
    #pragma unroll
    for (int r = 0; r < 8; ++r) {
      const int atom = base + r + half * 8;
      out[atom] = eacc[r] + b1[itr[r]];
    }
  }
}

// ---------------------------------------------------------------------------
extern "C" void kernel_launch(void* const* d_in, const int* in_sizes, int n_in,
                              void* d_out, int out_size, void* d_ws, size_t ws_size,
                              hipStream_t stream) {
  const float* rij      = (const float*)d_in[0];
  const float* unitv    = (const float*)d_in[1];
  const int*   itype    = (const int*)d_in[2];
  const int*   jtype    = (const int*)d_in[3];
  const float* c2       = (const float*)d_in[4];
  const float* c3       = (const float*)d_in[5];
  const float* q_scaler = (const float*)d_in[6];
  const float* W0       = (const float*)d_in[7];
  const float* b0       = (const float*)d_in[8];
  const float* W1       = (const float*)d_in[9];
  const float* b1       = (const float*)d_in[10];
  float* out = (float*)d_out;

  char* ws = (char*)d_ws;
  _Float16* featH = (_Float16*)ws;                                   // 8192*32*2   = 524288 B
  _Float16* W0p   = (_Float16*)(ws + 524288);                        // 7168*2      =  14336 B
  float*    b0p   = (float*)   (ws + 524288 + 14336);                // 224*4       =    896 B
  float*    W1p   = (float*)   (ws + 524288 + 14336 + 896);          // 224*4       =    896 B

  nep_feat_kernel<<<N_ATOMS / 8, 256, 0, stream>>>(
      rij, unitv, itype, jtype, c2, c3, q_scaler, featH);
  nep_repack_kernel<<<(2 * NTILES * 32 * 16 + 255) / 256, 256, 0, stream>>>(
      W0, b0, W1, W0p, b0p, W1p);
  nep_mlp_kernel<<<N_ATOMS / 16 / 8, 256, 0, stream>>>(
      featH, W0p, itype, b0p, W1p, b1, out);
}
